// GRULayer_45226005627379
// MI455X (gfx1250) — matrix operand order repaired and checked by reading, not compiled, not run on hardware
//
#include <hip/hip_runtime.h>
#include <hip/hip_bf16.h>

// GRU on MI455X (gfx1250): bf16 WMMA (v_wmma_f32_16x16x32_bf16) for both the
// input projection (fused per-step, since gi is single-use) and the recurrent
// matmul. One kernel launch per timestep provides the sequential dependency.

#define T_ 512
#define B_ 256
#define D_ 256
#define H_ 512

typedef __attribute__((ext_vector_type(16))) __bf16 v16bf;
typedef __attribute__((ext_vector_type(8)))  float  v8f;

__device__ __forceinline__ unsigned short f2bf(float x) {
    unsigned int u = __float_as_uint(x);
    unsigned int r = u + 0x7FFFu + ((u >> 16) & 1u);   // round-to-nearest-even
    return (unsigned short)(r >> 16);
}

__device__ __forceinline__ v8f wmma_bf16(v16bf a, v16bf b, v8f c) {
    return __builtin_amdgcn_wmma_f32_16x16x32_bf16(
        /*neg_a=*/false, a, /*neg_b=*/false, b,
        /*c_mod=*/(short)0, c, /*reuse_a=*/false, /*reuse_b=*/false);
}

// ---- prep: f32 -> bf16 bulk convert (for x) --------------------------------
__global__ __launch_bounds__(256) void convert_f32_bf16(
    const float* __restrict__ src, unsigned short* __restrict__ dst, int n) {
    int i = blockIdx.x * 256 + threadIdx.x;
    if (i < n) dst[i] = f2bf(src[i]);
}

// ---- prep: repack weight (3H x K, row major f32) into WMMA B fragments -----
// Fragment layout (B operand; contiguous k, same order as loadA): lane = (n%16) + 16*(kk/16),
// halfs[i] = K = kt*32 + (kk/16)*16 + i  -> 16 contiguous K values per lane.
// Stored lane-major so each lane loads its 32B fragment with 2x b128.
__global__ __launch_bounds__(256) void pack_w(
    const float* __restrict__ w, unsigned short* __restrict__ dst, int K) {
    int idx = blockIdx.x * 256 + threadIdx.x;
    int total = 3 * H_ * K;
    if (idx >= total) return;
    int n = idx / K;          // row of w (gate-major)
    int k = idx % K;
    int g  = n / H_;
    int j  = n % H_;          // column within gate (the WMMA N dimension)
    int nt = j >> 4;          // N tile
    int nl = j & 15;
    int kt = k >> 5;          // K tile (K=32 per WMMA)
    int kk = k & 31;
    int lane = nl + ((kk >> 4) << 4);
    int i    = kk & 15;
    int Ktiles = K >> 5;
    int NT = H_ >> 4;         // 32 tiles per gate
    size_t o = (((((size_t)g * NT + nt) * Ktiles + kt) * 32) + lane) * 16 + i;
    dst[o] = f2bf(w[idx]);
}

// ---- fragment loaders ------------------------------------------------------
#define H_ROW 520   // (512 + 8) halfs: +16B/row rotates LDS banks
#define X_ROW 264   // (256 + 8) halfs

// A operand from LDS, contiguous-k order (matches pack_w): lane m = lane&15,
// halfs[0..15]=K k0..k0+15, k0 = kt*32+((lane>>4)<<4)
__device__ __forceinline__ v16bf loadA(const unsigned short* s, int rowStride,
                                       int lane, int kt) {
    int m  = lane & 15;
    int k0 = (kt << 5) + ((lane >> 4) << 4);
    const unsigned short* p = s + m * rowStride + k0;
    union { v16bf v; uint4 u[2]; } u;
    u.u[0] = *reinterpret_cast<const uint4*>(p);        // ds_read_b128
    u.u[1] = *reinterpret_cast<const uint4*>(p + 8);    // ds_read_b128
    return u.v;
}

__device__ __forceinline__ v16bf loadB(const unsigned short* __restrict__ pack,
                                       int g, int nt, int kt, int Ktiles, int lane) {
    const unsigned short* p =
        pack + (((((size_t)g * 32 + nt) * Ktiles + kt) * 32) + lane) * 16;
    return *reinterpret_cast<const v16bf*>(p);          // 2x global_load_b128
}

// ---- one GRU timestep ------------------------------------------------------
// grid = (B/16, (H/16)/8), block = 256 (8 waves); wave -> one 16x16 h-tile.
__global__ __launch_bounds__(256) void gru_step(
    const unsigned short* __restrict__ xt,    // B x D bf16 (this timestep)
    const int*            __restrict__ es,    // B  episode_starts[t]
    const float*          __restrict__ hprev, // B x H f32 (= outputs[t-1] or h0)
    float*                __restrict__ hout,  // B x H f32 (= outputs[t])
    const unsigned short* __restrict__ wih_p, // packed bf16, Ktiles=8
    const unsigned short* __restrict__ whh_p, // packed bf16, Ktiles=16
    const float* __restrict__ b_ih, const float* __restrict__ b_hh) {

    __shared__ unsigned short sh_h[16 * H_ROW];
    __shared__ unsigned short sh_x[16 * X_ROW];

    const int tid  = threadIdx.x;
    const int m0   = blockIdx.x * 16;

    // Stage keep*h (f32 -> bf16) and x[t] (bf16) tiles into LDS.
    for (int i = tid; i < 16 * H_; i += 256) {
        int r = i >> 9, c = i & (H_ - 1);
        float keep = es[m0 + r] ? 0.f : 1.f;
        sh_h[r * H_ROW + c] = f2bf(hprev[(size_t)(m0 + r) * H_ + c] * keep);
    }
    for (int i = tid; i < 16 * D_; i += 256) {
        int r = i >> 8, c = i & (D_ - 1);
        sh_x[r * X_ROW + c] = xt[(size_t)(m0 + r) * D_ + c];
    }
    __syncthreads();

    const int wave = tid >> 5;
    const int lane = tid & 31;
    const int nt   = blockIdx.y * 8 + wave;   // 0..31
    const int n0   = nt << 4;

    v8f a_r = {}, a_z = {}, a_gin = {}, a_ghn = {};

    // gi = x[t] @ w_ih^T : K = 256 -> 8 k-tiles, 3 gates
    #pragma unroll
    for (int kt = 0; kt < D_ / 32; ++kt) {
        v16bf a  = loadA(sh_x, X_ROW, lane, kt);
        v16bf br = loadB(wih_p, 0, nt, kt, D_ / 32, lane);
        v16bf bz = loadB(wih_p, 1, nt, kt, D_ / 32, lane);
        v16bf bn = loadB(wih_p, 2, nt, kt, D_ / 32, lane);
        a_r   = wmma_bf16(a, br, a_r);
        a_z   = wmma_bf16(a, bz, a_z);
        a_gin = wmma_bf16(a, bn, a_gin);
    }
    // gh = (keep*h) @ w_hh^T : K = 512 -> 16 k-tiles, 3 gates
    #pragma unroll
    for (int kt = 0; kt < H_ / 32; ++kt) {
        v16bf a  = loadA(sh_h, H_ROW, lane, kt);
        v16bf br = loadB(whh_p, 0, nt, kt, H_ / 32, lane);
        v16bf bz = loadB(whh_p, 1, nt, kt, H_ / 32, lane);
        v16bf bn = loadB(whh_p, 2, nt, kt, H_ / 32, lane);
        a_r   = wmma_bf16(a, br, a_r);
        a_z   = wmma_bf16(a, bz, a_z);
        a_ghn = wmma_bf16(a, bn, a_ghn);
    }

    // Epilogue. C/D layout: lanes 0-15 -> M=v, N=lane; lanes 16-31 -> M=8+v.
    const int col = lane & 15;
    const int n   = n0 + col;
    const float br_b = b_ih[n]          + b_hh[n];
    const float bz_b = b_ih[H_ + n]     + b_hh[H_ + n];
    const float bin  = b_ih[2 * H_ + n];
    const float bhn  = b_hh[2 * H_ + n];
    const int rbase = m0 + ((lane >> 4) << 3);

    #pragma unroll
    for (int v = 0; v < 8; ++v) {
        int row = rbase + v;
        float keep = es[row] ? 0.f : 1.f;
        float hv   = hprev[(size_t)row * H_ + n] * keep;   // exact f32 carry path
        float pr = a_r[v] + br_b;
        float pz = a_z[v] + bz_b;
        float r  = 1.f / (1.f + __expf(-pr));
        float z  = 1.f / (1.f + __expf(-pz));
        float pn = (a_gin[v] + bin) + r * (a_ghn[v] + bhn);
        float e2 = __expf(2.f * pn);
        float th = 1.f - 2.f / (e2 + 1.f);                 // tanh
        hout[(size_t)row * H_ + n] = (1.f - z) * th + z * hv;
    }
}

extern "C" void kernel_launch(void* const* d_in, const int* in_sizes, int n_in,
                              void* d_out, int out_size, void* d_ws, size_t ws_size,
                              hipStream_t stream) {
    (void)in_sizes; (void)n_in; (void)out_size; (void)ws_size;
    const float* x    = (const float*)d_in[0];
    const float* h0   = (const float*)d_in[1];   // (1,B,H) zeros
    const int*   es   = (const int*)  d_in[2];
    const float* w_ih = (const float*)d_in[3];
    const float* w_hh = (const float*)d_in[4];
    const float* b_ih = (const float*)d_in[5];
    const float* b_hh = (const float*)d_in[6];
    float* out = (float*)d_out;                  // [T*B*H outputs | B*H h_final]

    char* ws = (char*)d_ws;
    unsigned short* x_bf  = (unsigned short*)ws;                         // T*B*D bf16
    unsigned short* wih_p = (unsigned short*)(ws + (size_t)T_ * B_ * D_ * 2);
    unsigned short* whh_p = wih_p + (size_t)3 * H_ * D_;

    convert_f32_bf16<<<(T_ * B_ * D_) / 256, 256, 0, stream>>>(
        x, x_bf, T_ * B_ * D_);
    pack_w<<<(3 * H_ * D_ + 255) / 256, 256, 0, stream>>>(w_ih, wih_p, D_);
    pack_w<<<(3 * H_ * H_ + 255) / 256, 256, 0, stream>>>(w_hh, whh_p, H_);

    for (int t = 0; t < T_; ++t) {
        const float* hprev = (t == 0) ? h0 : out + (size_t)(t - 1) * B_ * H_;
        gru_step<<<dim3(B_ / 16, (H_ / 16) / 8), 256, 0, stream>>>(
            x_bf + (size_t)t * B_ * D_, es + (size_t)t * B_, hprev,
            out + (size_t)t * B_ * H_, wih_p, whh_p, b_ih, b_hh);
    }
    // h_final = outputs[T-1]
    hipMemcpyAsync(out + (size_t)T_ * B_ * H_, out + (size_t)(T_ - 1) * B_ * H_,
                   (size_t)B_ * H_ * sizeof(float), hipMemcpyDeviceToDevice, stream);
}
